// GAT_62096637166285
// MI455X (gfx1250) — matrix operand order repaired and checked by reading, not compiled, not run on hardware
//
#include <hip/hip_runtime.h>

#define NN    8192
#define INCH  1024
#define NHID  512

typedef __attribute__((ext_vector_type(16))) __bf16 v16bf;
typedef __attribute__((ext_vector_type(8)))  __bf16 v8bf;
typedef __attribute__((ext_vector_type(8)))  float  v8f;

static __device__ __forceinline__ __bf16 f32_to_bf16(float f) {
  unsigned u = __builtin_bit_cast(unsigned, f);
  u += 0x7fffu + ((u >> 16) & 1u);          // round-to-nearest-even
  unsigned short s = (unsigned short)(u >> 16);
  return __builtin_bit_cast(__bf16, s);
}

// ---------------------------------------------------------------------------
// Vectorized fp32 -> bf16 (n must be a multiple of 4)
// ---------------------------------------------------------------------------
__global__ void cvt_f32_bf16_kernel(const float* __restrict__ in,
                                    __bf16* __restrict__ out, int n) {
  int i = (blockIdx.x * blockDim.x + threadIdx.x) * 4;
  if (i < n) {
    float4 f = *reinterpret_cast<const float4*>(in + i);
    out[i + 0] = f32_to_bf16(f.x);
    out[i + 1] = f32_to_bf16(f.y);
    out[i + 2] = f32_to_bf16(f.z);
    out[i + 3] = f32_to_bf16(f.w);
  }
}

// ---------------------------------------------------------------------------
// Register-blocked WMMA GEMM:
//   C[M][N] = A[M][K](bf16) x B[N][K](bf16)^T (+ bias)
// One wave computes a 64x64 output tile = 4x4 grid of 16x16 WMMA tiles.
// Per 32-wide k-step: 4 A frags + 4 B frags (16 x global_load_b128) -> 16 WMMA.
// Optional outputs: C (f32), Cb (bf16 row-major), CT (bf16 transposed [N][M]).
// ---------------------------------------------------------------------------
__global__ __launch_bounds__(256)
void gemm_bf16_wmma_kernel(const __bf16* __restrict__ A,
                           const __bf16* __restrict__ B,
                           const float* __restrict__ bias,
                           float* __restrict__ C,
                           __bf16* __restrict__ Cb,
                           __bf16* __restrict__ CT,
                           int M, int N, int K) {
  const int lane    = threadIdx.x & 31;
  const int wid     = blockIdx.x * (blockDim.x >> 5) + (threadIdx.x >> 5);
  const int nstrips = N >> 6;                 // 64-column strips
  const int mstrip  = wid / nstrips;
  const int nstrip  = wid - mstrip * nstrips;
  const int nloc    = lane & 15;
  const int half    = lane >> 4;

  // Per-lane row pointers (ISA fragment layouts, §7.12.2):
  //  A lane: row = tile_m + (lane&15); needs K chunks [kh*8,+8) and [kh*8+16,+8)
  //  B lane: col = tile_n + (lane&15); needs K chunks [kh*8,+8) and [kh*8+16,+8)
  const __bf16* Arow[4];
  const __bf16* Brow[4];
#pragma unroll
  for (int mi = 0; mi < 4; ++mi)
    Arow[mi] = A + (size_t)(mstrip * 64 + mi * 16 + nloc) * K;
#pragma unroll
  for (int ni = 0; ni < 4; ++ni)
    Brow[ni] = B + (size_t)(nstrip * 64 + ni * 16 + nloc) * K;

  v8f acc[4][4] = {};

  for (int k = 0; k < K; k += 32) {
    const int ka = k + half * 8;    // A: two 8-elem chunks, 16 apart
    const int kb = k + half * 8;    // B: two 8-elem chunks, 16 apart

    union F { v16bf v; v8bf h[2]; };
    F a[4], b[4];
#pragma unroll
    for (int mi = 0; mi < 4; ++mi) {
      a[mi].h[0] = *reinterpret_cast<const v8bf*>(Arow[mi] + ka);
      a[mi].h[1] = *reinterpret_cast<const v8bf*>(Arow[mi] + ka + 16);
    }
#pragma unroll
    for (int ni = 0; ni < 4; ++ni) {
      b[ni].h[0] = *reinterpret_cast<const v8bf*>(Brow[ni] + kb);
      b[ni].h[1] = *reinterpret_cast<const v8bf*>(Brow[ni] + kb + 16);
    }

    if (k + 32 < K) {               // prefetch the streamed A matrix
#pragma unroll
      for (int mi = 0; mi < 4; ++mi)
        __builtin_prefetch(Arow[mi] + ka + 32, 0, 1);
    }

#pragma unroll
    for (int mi = 0; mi < 4; ++mi)
#pragma unroll
      for (int ni = 0; ni < 4; ++ni)
        acc[mi][ni] = __builtin_amdgcn_wmma_f32_16x16x32_bf16(
            false, a[mi].v, false, b[ni].v,
            (short)0, acc[mi][ni], false, false);
  }

  // ---- Epilogue ----
  float bn[4];
#pragma unroll
  for (int ni = 0; ni < 4; ++ni)
    bn[ni] = bias ? bias[nstrip * 64 + ni * 16 + nloc] : 0.0f;

#pragma unroll
  for (int mi = 0; mi < 4; ++mi) {
#pragma unroll
    for (int i = 0; i < 8; ++i) {
      const int row = mstrip * 64 + mi * 16 + half * 8 + i;  // D: VGPR i -> row i+8*half
#pragma unroll
      for (int ni = 0; ni < 4; ++ni) {
        const int col = nstrip * 64 + ni * 16 + nloc;
        const float v = acc[mi][ni][i] + bn[ni];
        if (C)  C [(size_t)row * N + col] = v;
        if (Cb) Cb[(size_t)row * N + col] = f32_to_bf16(v);
        if (CT) CT[(size_t)col * M + row] = f32_to_bf16(v);
      }
    }
  }
}

// ---------------------------------------------------------------------------
// s1[i] = sum_j tanh(h1[i,j]) * a_w[j] ;  s2[i] = sum_j tanh(h2[i,j]) * a_w[F+j]
// ---------------------------------------------------------------------------
__global__ __launch_bounds__(128)
void score_kernel(const float* __restrict__ h1, const float* __restrict__ h2,
                  const float* __restrict__ a_w,
                  float* __restrict__ s1, float* __restrict__ s2) {
  __shared__ float r1[128];
  __shared__ float r2[128];
  const int row = blockIdx.x;
  const int t = threadIdx.x;
  float acc1 = 0.f, acc2 = 0.f;
  const float* h1r = h1 + (size_t)row * NHID;
  const float* h2r = h2 + (size_t)row * NHID;
  for (int j = t; j < NHID; j += 128) {
    acc1 += tanhf(h1r[j]) * a_w[j];
    acc2 += tanhf(h2r[j]) * a_w[NHID + j];
  }
  r1[t] = acc1; r2[t] = acc2;
  __syncthreads();
  for (int s = 64; s > 0; s >>= 1) {
    if (t < s) { r1[t] += r1[t + s]; r2[t] += r2[t + s]; }
    __syncthreads();
  }
  if (t == 0) { s1[row] = r1[0]; s2[row] = r2[0]; }
}

// ---------------------------------------------------------------------------
// Row softmax with adjacency mask. One block per row; the e-row is staged in
// 32 KB LDS so the 268 MB adj matrix is read exactly once. Writes both the
// fp32 attn (model output) and a bf16 copy (A-matrix of the final GEMM).
// ---------------------------------------------------------------------------
__global__ __launch_bounds__(256)
void softmax_kernel(const int* __restrict__ adj,
                    const float* __restrict__ s1, const float* __restrict__ s2,
                    const float* __restrict__ a_b,
                    float* __restrict__ attn, __bf16* __restrict__ attn_b) {
  __shared__ float ev[NN];
  __shared__ float red[256];
  const int row = blockIdx.x;
  const int t = threadIdx.x;
  const float ab = a_b[0];
  const float s1v = s1[row];
  const int* arow = adj + (size_t)row * NN;

  float mx = -3.0e38f;
  for (int j = t; j < NN; j += 256) {
    float e = s1v + s2[j] + ab;
    e = (e > 0.f) ? e : 0.2f * e;            // LeakyReLU(0.2)
    e = (arow[j] > 0) ? e : -9.0e15f;        // adjacency mask
    ev[j] = e;
    mx = fmaxf(mx, e);
  }
  red[t] = mx;
  __syncthreads();
  for (int s = 128; s > 0; s >>= 1) {
    if (t < s) red[t] = fmaxf(red[t], red[t + s]);
    __syncthreads();
  }
  mx = red[0];
  __syncthreads();

  float sum = 0.f;
  for (int j = t; j < NN; j += 256) {
    float p = __expf(ev[j] - mx);
    ev[j] = p;
    sum += p;
  }
  red[t] = sum;
  __syncthreads();
  for (int s = 128; s > 0; s >>= 1) {
    if (t < s) red[t] += red[t + s];
    __syncthreads();
  }
  const float inv = 1.0f / red[0];
  float*  orow  = attn   + (size_t)row * NN;
  __bf16* orowb = attn_b + (size_t)row * NN;
  for (int j = t; j < NN; j += 256) {
    const float p = ev[j] * inv;
    orow[j]  = p;
    orowb[j] = f32_to_bf16(p);
  }
}

// ---------------------------------------------------------------------------
extern "C" void kernel_launch(void* const* d_in, const int* in_sizes, int n_in,
                              void* d_out, int out_size, void* d_ws, size_t ws_size,
                              hipStream_t stream) {
  (void)in_sizes; (void)n_in; (void)out_size; (void)ws_size;
  const float* x    = (const float*)d_in[0];
  const int*   adj  = (const int*)  d_in[1];
  const float* W_em = (const float*)d_in[2];
  const float* b_em = (const float*)d_in[3];
  const float* W1   = (const float*)d_in[4];
  const float* b1   = (const float*)d_in[5];
  const float* W2   = (const float*)d_in[6];
  const float* b2   = (const float*)d_in[7];
  const float* a_w  = (const float*)d_in[8];
  const float* a_b  = (const float*)d_in[9];

  float* h_prime = (float*)d_out;                       // [8192, 512]
  float* attn    = h_prime + (size_t)NN * NHID;         // [8192, 8192]

  char* ws = (char*)d_ws;
  size_t off = 0;
  auto take = [&](size_t bytes) -> void* {
    void* p = ws + off;
    off += (bytes + 255) & ~(size_t)255;
    return p;
  };
  __bf16* x_b    = (__bf16*)take((size_t)NN * INCH * 2);    // 16 MB
  __bf16* Wem_b  = (__bf16*)take((size_t)NHID * INCH * 2);
  __bf16* W1_b   = (__bf16*)take((size_t)NHID * NHID * 2);
  __bf16* W2_b   = (__bf16*)take((size_t)NHID * NHID * 2);
  __bf16* h_b    = (__bf16*)take((size_t)NN * NHID * 2);    // h, row-major bf16
  __bf16* hT_b   = (__bf16*)take((size_t)NHID * NN * 2);    // h^T bf16
  float*  h1     = (float*) take((size_t)NN * NHID * 4);
  float*  h2     = (float*) take((size_t)NN * NHID * 4);
  float*  s1     = (float*) take((size_t)NN * 4);
  float*  s2     = (float*) take((size_t)NN * 4);
  __bf16* attn_b = (__bf16*)take((size_t)NN * NN * 2);      // 128 MB

  // One-shot conversions
  cvt_f32_bf16_kernel<<<(NN * INCH / 4 + 255) / 256, 256, 0, stream>>>(x, x_b, NN * INCH);
  cvt_f32_bf16_kernel<<<(NHID * INCH / 4 + 255) / 256, 256, 0, stream>>>(W_em, Wem_b, NHID * INCH);
  cvt_f32_bf16_kernel<<<(NHID * NHID / 4 + 255) / 256, 256, 0, stream>>>(W1, W1_b, NHID * NHID);
  cvt_f32_bf16_kernel<<<(NHID * NHID / 4 + 255) / 256, 256, 0, stream>>>(W2, W2_b, NHID * NHID);

  const int waves  = (NN / 64) * (NHID / 64);   // 1024 wave-tiles
  const int blocks = waves / 8;                 // 8 waves per 256-thread block

  // h = x @ W_em^T + b_em   -> emit bf16 h (row-major) and bf16 h^T
  gemm_bf16_wmma_kernel<<<blocks, 256, 0, stream>>>(
      x_b, Wem_b, b_em, nullptr, h_b, hT_b, NN, NHID, INCH);
  // h1 = h @ W1^T + b1 ; h2 = h @ W2^T + b2   (fp32 outputs for tanh scores)
  gemm_bf16_wmma_kernel<<<blocks, 256, 0, stream>>>(
      h_b, W1_b, b1, h1, nullptr, nullptr, NN, NHID, NHID);
  gemm_bf16_wmma_kernel<<<blocks, 256, 0, stream>>>(
      h_b, W2_b, b2, h2, nullptr, nullptr, NN, NHID, NHID);

  // per-node attention scores
  score_kernel<<<NN, 128, 0, stream>>>(h1, h2, a_w, s1, s2);
  // masked row softmax -> attn fp32 (output) + attn bf16 (for final GEMM)
  softmax_kernel<<<NN, 256, 0, stream>>>(adj, s1, s2, a_b, attn, attn_b);

  // h_prime = attn @ h
  gemm_bf16_wmma_kernel<<<blocks, 256, 0, stream>>>(
      attn_b, hT_b, nullptr, h_prime, nullptr, nullptr, NN, NHID, NN);
}
